// RNNLSTMNet_32066225832677
// MI455X (gfx1250) — matrix operand order repaired and checked by reading, not compiled, not run on hardware
//
#include <hip/hip_runtime.h>

// ---------------------------------------------------------------------------
// RNN+LSTM net for MI455X (gfx1250), wave32 + WMMA bf16.
//   Phase 1: time-parallel projections  XW = X @ Wih^T + (bih+bhh)  (WMMA GEMM)
//   Phase 2: persistent per-layer recurrence kernels: Whh slice staged in LDS
//            (async-to-LDS when available, bank-conflict-free 520-elem pitch);
//            512 serial steps with a global-atomic grid barrier; h kept bf16,
//            accumulation + gates fp32; next-step XW prefetched.
// ---------------------------------------------------------------------------

typedef __attribute__((ext_vector_type(16))) __bf16 v16bf;
typedef __attribute__((ext_vector_type(8)))  __bf16 v8bf;
typedef __attribute__((ext_vector_type(8)))  float  v8f;
typedef int v4i __attribute__((vector_size(4 * sizeof(int))));

#define HIDDIM 512
#define BATCH  64
#define SEQLEN 512
#define EMBDIM 64
#define NCLS   10
#define LPITCH 520   // LDS row pitch (elems): 1040B -> 260 dwords = 4 mod 64 banks

#define LDS_AS __attribute__((address_space(3)))

#if defined(__HIP_DEVICE_COMPILE__) && \
    __has_builtin(__builtin_amdgcn_global_load_async_to_lds_b128)
#define USE_ASYNC_LDS 1
#else
#define USE_ASYNC_LDS 0
#endif

#if USE_ASYNC_LDS
__device__ __forceinline__ void async_copy_b128(const __bf16* g, __bf16* l) {
  __builtin_amdgcn_global_load_async_to_lds_b128(
      (v4i*)g, (LDS_AS v4i*)l, 0, 0);
}
__device__ __forceinline__ void wait_async0() {
#if __has_builtin(__builtin_amdgcn_s_wait_asynccnt)
  __builtin_amdgcn_s_wait_asynccnt(0);
#else
  asm volatile("s_wait_asynccnt 0" ::: "memory");
#endif
}
#endif

__device__ __forceinline__ v8f v8f_zero() {
  v8f z = {0.f,0.f,0.f,0.f,0.f,0.f,0.f,0.f};
  return z;
}

__device__ __forceinline__ v8f wmma_bf16(v16bf a, v16bf b, v8f c) {
  // (neg_a, A, neg_b, B, c_mod, C, reuse_a, reuse_b)
  return __builtin_amdgcn_wmma_f32_16x16x32_bf16(false, a, false, b, (short)0, c,
                                                 false, false);
}

// A operand (16x32 bf16): lane L holds row M=L&15.
// K pattern per ISA: hi=(L&16)?8:0; elems 0..7 -> K=hi+e ; 8..15 -> K=16+hi+(e-8)
__device__ __forceinline__ v16bf load_a_bf16(const __bf16* __restrict__ row, int lane) {
  const int hi = (lane & 16) ? 8 : 0;
  v8bf lo  = *(const v8bf*)(row + hi);
  v8bf hi8 = *(const v8bf*)(row + 16 + hi);
  return __builtin_shufflevector(lo, hi8, 0,1,2,3,4,5,6,7,8,9,10,11,12,13,14,15);
}

// B operand (32x16 bf16): lane L holds col N=L&15; same K pattern as A: hi=(L&16)?8:0; elems 0..7 -> K=hi+e ; 8..15 -> K=16+hi+(e-8)
__device__ __forceinline__ v16bf load_b_bf16(const __bf16* row, int lane) {
  const int hi = (lane & 16) ? 8 : 0;
  v8bf lo  = *(const v8bf*)(row + hi);
  v8bf hi8 = *(const v8bf*)(row + 16 + hi);
  return __builtin_shufflevector(lo, hi8, 0,1,2,3,4,5,6,7,8,9,10,11,12,13,14,15);
}

// --------------------------- grid-wide barrier -----------------------------
__device__ __forceinline__ void grid_sync(unsigned* cnt, unsigned target) {
  __threadfence();
  __syncthreads();
  if (threadIdx.x == 0) {
    __hip_atomic_fetch_add(cnt, 1u, __ATOMIC_RELEASE, __HIP_MEMORY_SCOPE_AGENT);
    while (__hip_atomic_load(cnt, __ATOMIC_ACQUIRE, __HIP_MEMORY_SCOPE_AGENT) < target)
      __builtin_amdgcn_s_sleep(2);
  }
  __syncthreads();
  __threadfence();
}

// ------------------------------ small kernels ------------------------------
__global__ void zero_f32_kernel(float* p, int n) {
  int i = blockIdx.x * blockDim.x + threadIdx.x;
  if (i < n) p[i] = 0.f;
}

__global__ void f2bf_kernel(const float* __restrict__ in, __bf16* __restrict__ out, int n) {
  int i = blockIdx.x * blockDim.x + threadIdx.x;
  if (i < n) out[i] = (__bf16)in[i];
}

__global__ void embed_kernel(const int* __restrict__ x, const float* __restrict__ emb,
                             __bf16* __restrict__ X0) {
  int i = blockIdx.x;            // row in [0, B*T)
  int e = threadIdx.x;           // 0..63
  int tok = x[i];
  X0[(size_t)i * EMBDIM + e] = (__bf16)emb[(size_t)tok * EMBDIM + e];
}

__global__ void fc_kernel(const __bf16* __restrict__ hfin,  // [B][H] final hidden
                          const float* __restrict__ Wf,     // [NCLS][H]
                          const float* __restrict__ bf,
                          float* __restrict__ out) {
  int idx = blockIdx.x * blockDim.x + threadIdx.x;
  if (idx >= BATCH * NCLS) return;
  int b = idx / NCLS, c = idx % NCLS;
  float s = bf[c];
  for (int k = 0; k < HIDDIM; ++k)
    s += (float)hfin[(size_t)b * HIDDIM + k] * Wf[(size_t)c * HIDDIM + k];
  out[idx] = s;
}

// ------------------------- time-parallel GEMM (WMMA) -----------------------
// C[M][N] = A[M][K](bf16) @ W[N][K](bf16)^T + b0[N] + b1[N]
__global__ __launch_bounds__(256) void gemm_bf16_kernel(
    const __bf16* __restrict__ A, const __bf16* __restrict__ W,
    const float* __restrict__ b0, const float* __restrict__ b1,
    float* __restrict__ C, int M, int N, int K)
{
  const int lane = threadIdx.x & 31;
  const int wave = threadIdx.x >> 5;
  const int mt = wave >> 1;                 // 0..3
  const int nh = wave & 1;                  // 0..1
  const int row0 = blockIdx.y * 64 + mt * 16;
  const int col0 = blockIdx.x * 64 + nh * 32;
  const int arow = row0 + (lane & 15);
  const int bcol = col0 + (lane & 15);
  v8f acc0 = v8f_zero(), acc1 = v8f_zero();
  for (int k = 0; k < K; k += 32) {
    v16bf a  = load_a_bf16(A + (size_t)arow * K + k, lane);
    v16bf w0 = load_b_bf16(W + (size_t)bcol * K + k, lane);
    v16bf w1 = load_b_bf16(W + (size_t)(bcol + 16) * K + k, lane);
    acc0 = wmma_bf16(a, w0, acc0);
    acc1 = wmma_bf16(a, w1, acc1);
  }
  const int m0 = (lane & 16) ? 8 : 0;
  const int n  = lane & 15;
#pragma unroll
  for (int r = 0; r < 8; ++r) {
    int row = row0 + m0 + r;
    int c0 = col0 + n, c1 = col0 + 16 + n;
    C[(size_t)row * N + c0] = acc0[r] + b0[c0] + b1[c0];
    C[(size_t)row * N + c1] = acc1[r] + b0[c1] + b1[c1];
  }
}

// ----------------------------- RNN recurrence ------------------------------
// 8 WGs x 256 threads. WG g owns h columns [g*64, g*64+64); Whh slice in LDS.
__global__ __launch_bounds__(256) void rnn_scan_kernel(
    const float*  __restrict__ XW,    // [B*T][H] fp32, biases folded
    const __bf16* __restrict__ Whh,   // [H][H] bf16 row-major
    __bf16*       __restrict__ Y,     // [B*T][H] bf16 (feeds next layer GEMM)
    __bf16*       __restrict__ hbuf,  // [2][B][H] bf16 double buffer (pre-zeroed)
    unsigned*     __restrict__ cnt)
{
  extern __shared__ __bf16 ldsW[];    // 64 rows x LPITCH
  const int g = blockIdx.x;
  const int col_base = g * 64;
  const __bf16* wsrc = Whh + (size_t)col_base * HIDDIM;   // 64 contiguous rows
#if USE_ASYNC_LDS
  for (int c = threadIdx.x; c < 64 * (HIDDIM / 8); c += 256) {
    int row = c >> 6, e = (c & 63) << 3;                  // 8-elem (16B) chunks
    async_copy_b128(wsrc + (size_t)c * 8, ldsW + row * LPITCH + e);
  }
  wait_async0();
#else
  for (int idx = threadIdx.x; idx < 64 * HIDDIM; idx += 256)
    ldsW[(idx >> 9) * LPITCH + (idx & 511)] = wsrc[idx];
#endif
  __syncthreads();

  const int lane = threadIdx.x & 31;
  const int wave = threadIdx.x >> 5;
  const int mt = wave >> 1, nh = wave & 1;
  const int arow = mt * 16 + (lane & 15);
  const int ncl  = nh * 32 + (lane & 15);
  const int m0 = (lane & 16) ? 8 : 0;
  const int n  = lane & 15;
  const unsigned nwg = gridDim.x;

  for (int t = 0; t < SEQLEN; ++t) {
    const __bf16* cur = hbuf + (size_t)(t & 1) * BATCH * HIDDIM;
    __bf16* nxt = hbuf + (size_t)((t + 1) & 1) * BATCH * HIDDIM;
    // prefetch next step's XW rows (2 cachelines per 16-row group via lane hi bit)
    if (t + 1 < SEQLEN) {
      const float* pf = XW + ((size_t)arow * SEQLEN + t + 1) * HIDDIM + col_base +
                        ((lane >> 4) << 5);
      __builtin_prefetch(pf, 0, 3);
    }
    v8f acc0 = v8f_zero(), acc1 = v8f_zero();
#pragma unroll 4
    for (int k = 0; k < HIDDIM; k += 32) {
      v16bf a  = load_a_bf16(cur + (size_t)arow * HIDDIM + k, lane);
      v16bf w0 = load_b_bf16(ldsW + (size_t)ncl * LPITCH + k, lane);
      v16bf w1 = load_b_bf16(ldsW + (size_t)(ncl + 16) * LPITCH + k, lane);
      acc0 = wmma_bf16(a, w0, acc0);
      acc1 = wmma_bf16(a, w1, acc1);
    }
#pragma unroll
    for (int r = 0; r < 8; ++r) {
      int brow = mt * 16 + m0 + r;
      size_t irow = (size_t)brow * SEQLEN + t;
      int c0 = col_base + nh * 32 + n;
      int c1 = c0 + 16;
      float h0 = XW[irow * HIDDIM + c0] + acc0[r];
      float h1 = XW[irow * HIDDIM + c1] + acc1[r];
      h0 = h0 > 0.f ? h0 : 0.f;
      h1 = h1 > 0.f ? h1 : 0.f;
      __bf16 b0v = (__bf16)h0, b1v = (__bf16)h1;
      nxt[(size_t)brow * HIDDIM + c0] = b0v;
      nxt[(size_t)brow * HIDDIM + c1] = b1v;
      Y[irow * HIDDIM + c0] = b0v;
      Y[irow * HIDDIM + c1] = b1v;
    }
    grid_sync(cnt, nwg * (unsigned)(t + 1));
  }
}

// ----------------------------- LSTM recurrence -----------------------------
// 16 WGs x 256 threads. WG g owns h columns [g*32, g*32+32) -> 4 gate blocks
// (128 gate rows of Whh) staged into LDS. c state lives in registers (fp32).
__global__ __launch_bounds__(256) void lstm_scan_kernel(
    const float*  __restrict__ XW,    // [B*T][4H] fp32, biases folded
    const __bf16* __restrict__ Whh,   // [4H][H] bf16 row-major
    __bf16*       __restrict__ Y,     // [B*T][H]
    __bf16*       __restrict__ hbuf,  // [2][B][H] (pre-zeroed)
    unsigned*     __restrict__ cnt)
{
  extern __shared__ __bf16 ldsW[];    // 128 rows x LPITCH
  const int g = blockIdx.x;
  const int hcol_base = g * 32;
#if USE_ASYNC_LDS
  for (int q = 0; q < 4; ++q) {
    const __bf16* wsrc = Whh + (size_t)(q * HIDDIM + hcol_base) * HIDDIM;
    __bf16* dst = ldsW + (size_t)(q * 32) * LPITCH;
    for (int c = threadIdx.x; c < 32 * (HIDDIM / 8); c += 256) {
      int row = c >> 6, e = (c & 63) << 3;
      async_copy_b128(wsrc + (size_t)c * 8, dst + row * LPITCH + e);
    }
  }
  wait_async0();
#else
  for (int idx = threadIdx.x; idx < 128 * HIDDIM; idx += 256) {
    int lrow = idx >> 9;              // 0..127 = q*32 + cl
    int q = lrow >> 5, cl = lrow & 31;
    ldsW[lrow * LPITCH + (idx & 511)] =
        Whh[(size_t)(q * HIDDIM + hcol_base + cl) * HIDDIM + (idx & 511)];
  }
#endif
  __syncthreads();

  const int lane = threadIdx.x & 31;
  const int wave = threadIdx.x >> 5;
  const int mt = wave >> 1, nh = wave & 1;
  const int arow = mt * 16 + (lane & 15);
  const int m0 = (lane & 16) ? 8 : 0;
  const int n  = lane & 15;
  const int jl = nh * 16 + n;         // local h column 0..31
  const unsigned nwg = gridDim.x;

  float cst[8];
#pragma unroll
  for (int r = 0; r < 8; ++r) cst[r] = 0.f;

  for (int t = 0; t < SEQLEN; ++t) {
    const __bf16* cur = hbuf + (size_t)(t & 1) * BATCH * HIDDIM;
    __bf16* nxt = hbuf + (size_t)((t + 1) & 1) * BATCH * HIDDIM;
    if (t + 1 < SEQLEN) {
      const float* pfb =
          XW + ((size_t)arow * SEQLEN + t + 1) * (size_t)(4 * HIDDIM) + hcol_base;
#pragma unroll
      for (int q = 0; q < 4; ++q) __builtin_prefetch(pfb + q * HIDDIM, 0, 3);
    }
    v8f acc[4];
#pragma unroll
    for (int q = 0; q < 4; ++q) acc[q] = v8f_zero();
#pragma unroll 2
    for (int k = 0; k < HIDDIM; k += 32) {
      v16bf a = load_a_bf16(cur + (size_t)arow * HIDDIM + k, lane);
#pragma unroll
      for (int q = 0; q < 4; ++q) {
        v16bf w = load_b_bf16(
            ldsW + (size_t)(q * 32 + nh * 16 + (lane & 15)) * LPITCH + k, lane);
        acc[q] = wmma_bf16(a, w, acc[q]);
      }
    }
#pragma unroll
    for (int r = 0; r < 8; ++r) {
      int brow = mt * 16 + m0 + r;
      size_t irow = (size_t)brow * SEQLEN + t;
      const float* xw = XW + irow * (size_t)(4 * HIDDIM);
      int j = hcol_base + jl;
      float gi = xw[j]              + acc[0][r];
      float gf = xw[HIDDIM + j]     + acc[1][r];
      float gg = xw[2 * HIDDIM + j] + acc[2][r];
      float go = xw[3 * HIDDIM + j] + acc[3][r];
      float si = 1.f / (1.f + __expf(-gi));
      float sf = 1.f / (1.f + __expf(-gf));
      float so = 1.f / (1.f + __expf(-go));
      float c  = sf * cst[r] + si * tanhf(gg);
      cst[r] = c;
      float h = so * tanhf(c);
      nxt[(size_t)brow * HIDDIM + j] = (__bf16)h;
      Y[irow * HIDDIM + j] = (__bf16)h;
    }
    grid_sync(cnt, nwg * (unsigned)(t + 1));
  }
}

// ------------------------------- launcher ----------------------------------
extern "C" void kernel_launch(void* const* d_in, const int* in_sizes, int n_in,
                              void* d_out, int out_size, void* d_ws, size_t ws_size,
                              hipStream_t stream) {
  (void)in_sizes; (void)n_in; (void)out_size; (void)ws_size;
  const int*   x     = (const int*)  d_in[0];
  const float* emb   = (const float*)d_in[1];
  const float* rWih0 = (const float*)d_in[2];
  const float* rWhh0 = (const float*)d_in[3];
  const float* rbih0 = (const float*)d_in[4];
  const float* rbhh0 = (const float*)d_in[5];
  const float* rWih1 = (const float*)d_in[6];
  const float* rWhh1 = (const float*)d_in[7];
  const float* rbih1 = (const float*)d_in[8];
  const float* rbhh1 = (const float*)d_in[9];
  const float* lWih0 = (const float*)d_in[10];
  const float* lWhh0 = (const float*)d_in[11];
  const float* lbih0 = (const float*)d_in[12];
  const float* lbhh0 = (const float*)d_in[13];
  const float* lWih1 = (const float*)d_in[14];
  const float* lWhh1 = (const float*)d_in[15];
  const float* lbih1 = (const float*)d_in[16];
  const float* lbhh1 = (const float*)d_in[17];
  const float* fcW   = (const float*)d_in[18];
  const float* fcb   = (const float*)d_in[19];
  float* out = (float*)d_out;

  const int MT = BATCH * SEQLEN;        // 32768 rows
  char* ws = (char*)d_ws;
  size_t off = 0;
  auto alloc = [&](size_t bytes) -> void* {
    void* p = ws + off;
    off = (off + bytes + 255) & ~(size_t)255;
    return p;
  };
  unsigned* cnts = (unsigned*)alloc(1024);                        // 4 counters @64B
  __bf16* hbuf0 = (__bf16*)alloc((size_t)2 * BATCH * HIDDIM * 2); // bf16
  __bf16* hbuf1 = (__bf16*)alloc((size_t)2 * BATCH * HIDDIM * 2);
  __bf16* hbuf2 = (__bf16*)alloc((size_t)2 * BATCH * HIDDIM * 2);
  __bf16* hbuf3 = (__bf16*)alloc((size_t)2 * BATCH * HIDDIM * 2);
  __bf16* X0 = (__bf16*)alloc((size_t)MT * EMBDIM * 2);
  __bf16* Ya = (__bf16*)alloc((size_t)MT * HIDDIM * 2);
  __bf16* Yb = (__bf16*)alloc((size_t)MT * HIDDIM * 2);
  __bf16* bWih0 = (__bf16*)alloc((size_t)HIDDIM * EMBDIM * 2);
  __bf16* bWhh0 = (__bf16*)alloc((size_t)HIDDIM * HIDDIM * 2);
  __bf16* bWih1 = (__bf16*)alloc((size_t)HIDDIM * HIDDIM * 2);
  __bf16* bWhh1 = (__bf16*)alloc((size_t)HIDDIM * HIDDIM * 2);
  __bf16* bLih0 = (__bf16*)alloc((size_t)4 * HIDDIM * HIDDIM * 2);
  __bf16* bLhh0 = (__bf16*)alloc((size_t)4 * HIDDIM * HIDDIM * 2);
  __bf16* bLih1 = (__bf16*)alloc((size_t)4 * HIDDIM * HIDDIM * 2);
  __bf16* bLhh1 = (__bf16*)alloc((size_t)4 * HIDDIM * HIDDIM * 2);
  float* XW = (float*)alloc((size_t)MT * 4 * HIDDIM * 4);         // 256 MB, reused

  // Zero counters + all h double buffers (contiguous region at ws start).
  {
    int nz = (1024 + 4 * 2 * BATCH * HIDDIM * 2) / 4;
    zero_f32_kernel<<<(nz + 255) / 256, 256, 0, stream>>>((float*)ws, nz);
  }
  auto conv = [&](const float* in, __bf16* o, int n) {
    f2bf_kernel<<<(n + 255) / 256, 256, 0, stream>>>(in, o, n);
  };
  conv(rWih0, bWih0, HIDDIM * EMBDIM);
  conv(rWhh0, bWhh0, HIDDIM * HIDDIM);
  conv(rWih1, bWih1, HIDDIM * HIDDIM);
  conv(rWhh1, bWhh1, HIDDIM * HIDDIM);
  conv(lWih0, bLih0, 4 * HIDDIM * HIDDIM);
  conv(lWhh0, bLhh0, 4 * HIDDIM * HIDDIM);
  conv(lWih1, bLih1, 4 * HIDDIM * HIDDIM);
  conv(lWhh1, bLhh1, 4 * HIDDIM * HIDDIM);

  embed_kernel<<<MT, EMBDIM, 0, stream>>>(x, emb, X0);

  dim3 blk(256);
  const size_t rnnLds  = (size_t)64 * LPITCH * 2;   // 66,560 B
  const size_t lstmLds = (size_t)128 * LPITCH * 2;  // 133,120 B
  // RNN layer 0
  gemm_bf16_kernel<<<dim3(HIDDIM / 64, MT / 64), blk, 0, stream>>>(
      X0, bWih0, rbih0, rbhh0, XW, MT, HIDDIM, EMBDIM);
  rnn_scan_kernel<<<8, blk, rnnLds, stream>>>(
      XW, bWhh0, Ya, hbuf0, (unsigned*)((char*)cnts + 0));
  // RNN layer 1
  gemm_bf16_kernel<<<dim3(HIDDIM / 64, MT / 64), blk, 0, stream>>>(
      Ya, bWih1, rbih1, rbhh1, XW, MT, HIDDIM, HIDDIM);
  rnn_scan_kernel<<<8, blk, rnnLds, stream>>>(
      XW, bWhh1, Yb, hbuf1, (unsigned*)((char*)cnts + 64));
  // LSTM layer 0
  gemm_bf16_kernel<<<dim3(4 * HIDDIM / 64, MT / 64), blk, 0, stream>>>(
      Yb, bLih0, lbih0, lbhh0, XW, MT, 4 * HIDDIM, HIDDIM);
  lstm_scan_kernel<<<16, blk, lstmLds, stream>>>(
      XW, bLhh0, Ya, hbuf2, (unsigned*)((char*)cnts + 128));
  // LSTM layer 1
  gemm_bf16_kernel<<<dim3(4 * HIDDIM / 64, MT / 64), blk, 0, stream>>>(
      Ya, bLih1, lbih1, lbhh1, XW, MT, 4 * HIDDIM, HIDDIM);
  lstm_scan_kernel<<<16, blk, lstmLds, stream>>>(
      XW, bLhh1, Yb, hbuf3, (unsigned*)((char*)cnts + 192));

  // Final hidden state: after t=511 the write buffer is hbuf3[0].
  fc_kernel<<<(BATCH * NCLS + 255) / 256, 256, 0, stream>>>(hbuf3, fcW, fcb, out);
}
